// ProtoNet_13005160972748
// MI455X (gfx1250) — matrix operand order repaired and checked by reading, not compiled, not run on hardware
//
#include <hip/hip_runtime.h>
#include <hip/hip_bf16.h>

typedef __attribute__((ext_vector_type(16))) __bf16          v16bf;
typedef __attribute__((ext_vector_type(16))) unsigned short  v16us;
typedef __attribute__((ext_vector_type(8)))  float           v8f;

#define NQ 16384
#define MP 1024
#define DD 1024

// ---- bf16 helpers ----------------------------------------------------------
__device__ __forceinline__ unsigned short bf16_rne(float x) {
    unsigned u = __builtin_bit_cast(unsigned, x);
    u += 0x7FFFu + ((u >> 16) & 1u);
    return (unsigned short)(u >> 16);
}
__device__ __forceinline__ float bf16_to_f32(unsigned short h) {
    return __builtin_bit_cast(float, (unsigned)h << 16);
}
__device__ __forceinline__ unsigned fbits(float x) {
    return __builtin_bit_cast(unsigned, x);
}
__device__ __forceinline__ float trunc_bf16_f32(float x) {
    return __builtin_bit_cast(float, fbits(x) & 0xFFFF0000u);
}

// ---------------------------------------------------------------------------
// Pass 1: split proto into bf16 hi/lo, pre-swizzled into this kernel's B-operand
// layout (per 16x32 K-chunk tile: lane = (n%16) + 16*(k>=16), slot = k%16,
// 16 contiguous bf16 per lane => two b128 loads per operand in the GEMM).
// Also compute p2[m] = ||proto_m||^2.
// ---------------------------------------------------------------------------
__global__ __launch_bounds__(256) void protonet_split_proto(
    const float* __restrict__ proto,
    unsigned short* __restrict__ ph,
    unsigned short* __restrict__ pl,
    float* __restrict__ p2) {
    const int m   = blockIdx.x;      // 0..1023
    const int tid = threadIdx.x;     // 0..255
    const int nt  = m >> 4;          // column tile (N/16)
    const int ml  = m & 15;
    const float* row = proto + (size_t)m * DD;
    float ss = 0.f;
    for (int d = tid; d < DD; d += 256) {
        float x = row[d];
        ss += x * x;
        unsigned short h = bf16_rne(x);
        unsigned short l = bf16_rne(x - bf16_to_f32(h));
        const int kc   = d >> 5;     // K-chunk of 32
        const int kin  = d & 31;
        const int lane = ml + ((kin >> 4) << 4);
        const size_t off = (((size_t)(kc * 64 + nt) * 32) + lane) * 16 + (kin & 15);
        ph[off] = h;
        pl[off] = l;
    }
    __shared__ float sred[256];
    sred[tid] = ss;
    __syncthreads();
    for (int s = 128; s > 0; s >>= 1) {
        if (tid < s) sred[tid] += sred[tid + s];
        __syncthreads();
    }
    if (tid == 0) p2[m] = sred[0];
}

// ---------------------------------------------------------------------------
// Pass 2: fused (2*Q@P^T - p2) + row softmax.
// Block = 256 threads = 8 waves; rows [r0, r0+32) x all 1024 cols.
// A chunk (32 rows x 32 K fp32) is loaded+split to bf16 hi/lo ONCE per block
// into double-buffered LDS in the WMMA A-operand swizzle; each wave reads its
// fragments with ds_load_b128. Wave w owns cols [w*128, (w+1)*128) as 8
// N-tiles x 2 row-tiles; q.p ~= qh.ph + qh.pl + ql.ph (3 wmma/tile/chunk).
// ---------------------------------------------------------------------------
__global__ __launch_bounds__(256) void protonet_gemm_softmax(
    const float* __restrict__ query,
    const unsigned short* __restrict__ ph,
    const unsigned short* __restrict__ pl,
    const float* __restrict__ p2,
    float* __restrict__ out) {
    const int tid  = threadIdx.x;
    const int lane = tid & 31;
    const int w    = tid >> 5;        // wave 0..7
    const int ln   = lane & 15;       // column-within-tile
    const int half = lane >> 4;       // M+8 selector for C rows
    const int r0   = blockIdx.x * 32; // row block

    // [buf][hi=0/lo=1][(rt*32+lane)*16 + slot]
    __shared__ unsigned short lsA[2][2][1024];
    __shared__ float redM[8][32];
    __shared__ float redS[8][32];

    // staging role: thread -> (row, 4 consecutive K)
    const int srow  = tid >> 3;          // 0..31
    const int skq   = (tid & 7) << 2;    // 0,4,...,28
    const int g     = skq >> 3;          // K-group of 8
    const int slane = (srow & 15) + ((skq >> 4) << 4);
    const int sslot = skq & 15;
    const int srt   = srow >> 4;
    const int sidx  = (srt * 32 + slane) * 16 + sslot;   // uint2-aligned (slot%4==0)
    const float* sgl = query + (size_t)(r0 + srow) * DD + skq;

    v8f acc[2][8];
#pragma unroll
    for (int rt = 0; rt < 2; ++rt)
#pragma unroll
        for (int t = 0; t < 8; ++t)
#pragma unroll
            for (int j = 0; j < 8; ++j) acc[rt][t][j] = 0.f;

#pragma unroll 1
    for (int kc = 0; kc < 32; ++kc) {
        const int buf = kc & 1;
        // ---- cooperative stage: load 4 fp32, truncation-split, perm-pack ---
        {
            if (kc < 31) __builtin_prefetch(sgl + kc * 32 + 32, 0, 0);
            const float4 f = *(const float4*)(sgl + kc * 32);
            const unsigned h01 = __builtin_amdgcn_perm(fbits(f.y), fbits(f.x), 0x07060302u);
            const unsigned h23 = __builtin_amdgcn_perm(fbits(f.w), fbits(f.z), 0x07060302u);
            const float l0 = f.x - trunc_bf16_f32(f.x);
            const float l1 = f.y - trunc_bf16_f32(f.y);
            const float l2 = f.z - trunc_bf16_f32(f.z);
            const float l3 = f.w - trunc_bf16_f32(f.w);
            const unsigned q01 = __builtin_amdgcn_perm(fbits(l1), fbits(l0), 0x07060302u);
            const unsigned q23 = __builtin_amdgcn_perm(fbits(l3), fbits(l2), 0x07060302u);
            *(uint2*)(&lsA[buf][0][sidx]) = uint2{h01, h23};
            *(uint2*)(&lsA[buf][1][sidx]) = uint2{q01, q23};
        }
        __syncthreads();

        // ---- per-wave A fragments from LDS ---------------------------------
        v16bf ah[2], al[2];
#pragma unroll
        for (int rt = 0; rt < 2; ++rt) {
            ah[rt] = *(const v16bf*)(&lsA[buf][0][(rt * 32 + lane) * 16]);
            al[rt] = *(const v16bf*)(&lsA[buf][1][(rt * 32 + lane) * 16]);
        }

        // ---- B operands (pre-swizzled) + WMMAs -----------------------------
        const size_t bbase = (((size_t)(kc * 64 + w * 8) * 32) + lane) * 16;
        const unsigned short* bh = ph + bbase;
        const unsigned short* bl = pl + bbase;
#pragma unroll
        for (int t = 0; t < 8; ++t) {
            const v16bf bhv = *(const v16bf*)(bh + (size_t)t * 512);
            const v16bf blv = *(const v16bf*)(bl + (size_t)t * 512);
#pragma unroll
            for (int rt = 0; rt < 2; ++rt) {
                acc[rt][t] = __builtin_amdgcn_wmma_f32_16x16x32_bf16(
                    false, ah[rt], false, bhv, (short)0, acc[rt][t], false, false);
                acc[rt][t] = __builtin_amdgcn_wmma_f32_16x16x32_bf16(
                    false, al[rt], false, bhv, (short)0, acc[rt][t], false, false);
                acc[rt][t] = __builtin_amdgcn_wmma_f32_16x16x32_bf16(
                    false, ah[rt], false, blv, (short)0, acc[rt][t], false, false);
            }
        }
    }

    // ---- fused softmax over the 1024 columns (logit = 2*qp - p2[m]) -------
    float p2v[8];
#pragma unroll
    for (int t = 0; t < 8; ++t) p2v[t] = p2[w * 128 + t * 16 + ln];

    // C/D layout: vgpr j -> row M = j + 8*half, N = ln (+ tile/wave offsets).
    float gmax[2][8], ginv[2][8];

#pragma unroll
    for (int rt = 0; rt < 2; ++rt)
#pragma unroll
        for (int j = 0; j < 8; ++j) {
            float mx = -__builtin_inff();
#pragma unroll
            for (int t = 0; t < 8; ++t)
                mx = fmaxf(mx, 2.f * acc[rt][t][j] - p2v[t]);
#pragma unroll
            for (int s = 1; s < 16; s <<= 1)
                mx = fmaxf(mx, __shfl_xor(mx, s, 32));
            if (ln == 0) redM[w][rt * 16 + half * 8 + j] = mx;
        }
    __syncthreads();
#pragma unroll
    for (int rt = 0; rt < 2; ++rt)
#pragma unroll
        for (int j = 0; j < 8; ++j) {
            const int r = rt * 16 + half * 8 + j;
            float mx = redM[0][r];
#pragma unroll
            for (int ww = 1; ww < 8; ++ww) mx = fmaxf(mx, redM[ww][r]);
            gmax[rt][j] = mx;
        }

#pragma unroll
    for (int rt = 0; rt < 2; ++rt)
#pragma unroll
        for (int j = 0; j < 8; ++j) {
            float s = 0.f;
#pragma unroll
            for (int t = 0; t < 8; ++t) {
                const float e = __expf(2.f * acc[rt][t][j] - p2v[t] - gmax[rt][j]);
                acc[rt][t][j] = e;
                s += e;
            }
#pragma unroll
            for (int st = 1; st < 16; st <<= 1)
                s += __shfl_xor(s, st, 32);
            if (ln == 0) redS[w][rt * 16 + half * 8 + j] = s;
        }
    __syncthreads();
#pragma unroll
    for (int rt = 0; rt < 2; ++rt)
#pragma unroll
        for (int j = 0; j < 8; ++j) {
            const int r = rt * 16 + half * 8 + j;
            float s = 0.f;
#pragma unroll
            for (int ww = 0; ww < 8; ++ww) s += redS[ww][r];
            ginv[rt][j] = 1.f / s;
        }

    // ---- store (lanes 0-15 contiguous columns -> coalesced) ----------------
#pragma unroll
    for (int rt = 0; rt < 2; ++rt)
#pragma unroll
        for (int j = 0; j < 8; ++j) {
            const size_t row = (size_t)r0 + rt * 16 + half * 8 + j;
            float* o = out + row * MP + w * 128 + ln;
#pragma unroll
            for (int t = 0; t < 8; ++t) o[t * 16] = acc[rt][t][j] * ginv[rt][j];
        }
}

// ---------------------------------------------------------------------------
extern "C" void kernel_launch(void* const* d_in, const int* in_sizes, int n_in,
                              void* d_out, int out_size, void* d_ws, size_t ws_size,
                              hipStream_t stream) {
    (void)in_sizes; (void)n_in; (void)out_size; (void)ws_size;
    const float* query = (const float*)d_in[0];   // [16384,1024] f32
    const float* proto = (const float*)d_in[1];   // [1024,1024]  f32
    float* out = (float*)d_out;                   // [16384,1024] f32

    // workspace: ph (2MB) | pl (2MB) | p2 (4KB)
    unsigned short* ph = (unsigned short*)d_ws;
    unsigned short* pl = ph + (size_t)MP * DD;
    float* p2 = (float*)(pl + (size_t)MP * DD);

    protonet_split_proto<<<MP, 256, 0, stream>>>(proto, ph, pl, p2);
    protonet_gemm_softmax<<<NQ / 32, 256, 0, stream>>>(query, ph, pl, p2, out);
}
